// LinearJointTSAttention_62783831933557
// MI455X (gfx1250) — matrix operand order repaired and checked by reading, not compiled, not run on hardware
//
#include <hip/hip_runtime.h>
#include <hip/hip_bf16.h>
#include <stdint.h>

// ---------------------------------------------------------------------------
// Types for CDNA5 WMMA (wave32, 16x16x32 bf16 -> f32)
// ---------------------------------------------------------------------------
typedef __bf16 bf16_t;
typedef bf16_t v16bf __attribute__((ext_vector_type(16)));
typedef float  v8f   __attribute__((ext_vector_type(8)));

struct Frag32 { uint4 a, b; };

static __device__ __forceinline__ v16bf frag_cast(uint4 a, uint4 b) {
  Frag32 f{a, b};
  return __builtin_bit_cast(v16bf, f);
}
static __device__ __forceinline__ unsigned short f2bf(float x) {
  unsigned u = __builtin_bit_cast(unsigned, x);
  u += 0x7FFFu + ((u >> 16) & 1u);               // round-to-nearest-even
  return (unsigned short)(u >> 16);
}
static __device__ __forceinline__ float bf2f(unsigned short h) {
  unsigned u = ((unsigned)h) << 16;
  return __builtin_bit_cast(float, u);
}
// Low 32 bits of a generic pointer to __shared__ = LDS byte offset.
static __device__ __forceinline__ unsigned lds_off(const void* p) {
  return (unsigned)(uintptr_t)p;
}

// 64 bytes per lane, global -> LDS via async DMA path (ASYNCcnt).
// INST_OFFSET applies to both the LDS destination and the global address.
#define ASYNC_CP64(ldsoff, gptr)                                              \
  asm volatile(                                                               \
      "global_load_async_to_lds_b128 %0, %1, off\n\t"                         \
      "global_load_async_to_lds_b128 %0, %1, off offset:16\n\t"               \
      "global_load_async_to_lds_b128 %0, %1, off offset:32\n\t"               \
      "global_load_async_to_lds_b128 %0, %1, off offset:48"                   \
      :: "v"(ldsoff), "v"((unsigned long long)(uintptr_t)(gptr))              \
      : "memory")

#define ASYNC_WAIT0() asm volatile("s_wait_asynccnt 0x0" ::: "memory")

#define C_DIM   768
#define HEADS   12
#define EHEAD   64
#define NTOK    1568            // num_frames * N = 8*196 (per batch-group)
#define MROWS   25088           // B*N = 128*196

// ---------------------------------------------------------------------------
// Kernel 1: fp32 -> bf16 convert
// ---------------------------------------------------------------------------
__global__ void cvt_f32_bf16(const float* __restrict__ in,
                             unsigned short* __restrict__ out, long long n) {
  long long i = (long long)blockIdx.x * blockDim.x + threadIdx.x;
  long long stride = (long long)gridDim.x * blockDim.x;
  for (; i < n; i += stride) out[i] = f2bf(in[i]);
}

// ---------------------------------------------------------------------------
// Kernel 2: qkv = x @ Wqkv^T  (M=25088, N=2304, K=768), fused relu+scale
//   128x128 block tile, 4 waves, each wave 64x64 (16 WMMA accumulators)
//   Double-buffered LDS fed by global_load_async_to_lds_b128.
// ---------------------------------------------------------------------------
__global__ __launch_bounds__(128) void gemm_qkv(
    const unsigned short* __restrict__ xb,
    const unsigned short* __restrict__ wb,
    unsigned short* __restrict__ qb,
    unsigned short* __restrict__ kb,
    unsigned short* __restrict__ vb) {
  __shared__ __align__(16) unsigned short lA[2][128 * 32];
  __shared__ __align__(16) unsigned short lB[2][128 * 32];
  const int tid  = threadIdx.x;
  const int wave = tid >> 5, lane = tid & 31;
  const int wm = wave >> 1, wn = wave & 1;
  const int half = lane >> 4, lrow = lane & 15;
  const int m0 = blockIdx.x * 128;
  const int n0 = blockIdx.y * 128;

  const unsigned short* gA = xb + (size_t)(m0 + tid) * C_DIM;   // this thread's row
  const unsigned short* gB = wb + (size_t)(n0 + tid) * C_DIM;
  const unsigned ldsA0 = lds_off(&lA[0][0]) + tid * 64;
  const unsigned ldsA1 = lds_off(&lA[1][0]) + tid * 64;
  const unsigned ldsB0 = lds_off(&lB[0][0]) + tid * 64;
  const unsigned ldsB1 = lds_off(&lB[1][0]) + tid * 64;

  v8f acc[4][4] = {};

  // prologue: fetch k-tile 0 into buffer 0
  ASYNC_CP64(ldsA0, gA);
  ASYNC_CP64(ldsB0, gB);

  const int NK = C_DIM / 32;   // 24
  for (int i = 0; i < NK; ++i) {
    const int cur = i & 1;
    ASYNC_WAIT0();             // my async loads for buffer `cur` complete
    __syncthreads();           // everyone's loads done; everyone done reading cur^1
    if (i + 1 < NK) {          // prefetch next k-tile into the other buffer
      const unsigned short* pa = gA + (i + 1) * 32;
      const unsigned short* pb = gB + (i + 1) * 32;
      ASYNC_CP64(cur ? ldsA0 : ldsA1, pa);
      ASYNC_CP64(cur ? ldsB0 : ldsB1, pb);
    }
    const unsigned short* bufA = lA[cur];
    const unsigned short* bufB = lB[cur];

    v16bf bfrag[4];
    #pragma unroll
    for (int nt = 0; nt < 4; ++nt) {
      const uint4* pb4 = (const uint4*)(bufB + (wn * 64 + nt * 16 + lrow) * 32);
      bfrag[nt] = frag_cast(pb4[half * 2], pb4[half * 2 + 1]);   // K contiguous
    }
    #pragma unroll
    for (int mt = 0; mt < 4; ++mt) {
      const uint4* pa4 = (const uint4*)(bufA + (wm * 64 + mt * 16 + lrow) * 32);
      v16bf afrag = frag_cast(pa4[half * 2], pa4[half * 2 + 1]); // K contiguous
      #pragma unroll
      for (int nt = 0; nt < 4; ++nt)
        acc[mt][nt] = __builtin_amdgcn_wmma_f32_16x16x32_bf16(
            false, afrag, false, bfrag[nt], (short)0, acc[mt][nt], false, false);
    }
  }

  const float scale = 0.125f;  // E^-0.5
  #pragma unroll
  for (int nt = 0; nt < 4; ++nt) {
    int c = n0 + wn * 64 + nt * 16 + lrow;      // [0, 2304)
    int part = c / C_DIM;                       // 0=q 1=k 2=v
    int cc = c - part * C_DIM;
    unsigned short* dst = (part == 0) ? qb : (part == 1) ? kb : vb;
    bool act = part < 2;
    #pragma unroll
    for (int mt = 0; mt < 4; ++mt) {
      int mbase = m0 + wm * 64 + mt * 16 + half * 8;
      v8f a = acc[mt][nt];
      #pragma unroll
      for (int r = 0; r < 8; ++r) {
        float v = a[r];
        if (act) v = fmaxf(v, 0.f) + scale;
        dst[(size_t)(mbase + r) * C_DIM + cc] = f2bf(v);
      }
    }
  }
}

// ---------------------------------------------------------------------------
// Kernel 3: per (b,h): kv^T[d][e] = sum_n k[n][e]*v[n][d]; ksum[e] = sum_n k[n][e]
// ---------------------------------------------------------------------------
__global__ __launch_bounds__(256) void kv_kernel(
    const unsigned short* __restrict__ kb,
    const unsigned short* __restrict__ vb,
    unsigned short* __restrict__ kvt,
    float* __restrict__ ksum) {
  __shared__ __align__(16) unsigned short kt[32 * 64];
  __shared__ __align__(16) unsigned short vt[32 * 64];
  const int bh = blockIdx.x;
  const int bb = bh / HEADS, h = bh % HEADS;
  const int t = threadIdx.x;
  const int d = t & 63, eg = t >> 6;            // thread owns (d, e in eg*16..+16)
  const int lr = t >> 3, lc = (t & 7) * 8;      // LDS load assignment

  float acc[16] = {};
  float ks = 0.f;

  for (int nc = 0; nc < NTOK / 32; ++nc) {
    __syncthreads();
    size_t gbase = ((size_t)bb * NTOK + nc * 32 + lr) * C_DIM + h * EHEAD + lc;
    *(uint4*)(kt + lr * 64 + lc) = *(const uint4*)(kb + gbase);
    *(uint4*)(vt + lr * 64 + lc) = *(const uint4*)(vb + gbase);
    __syncthreads();
    #pragma unroll 4
    for (int n = 0; n < 32; ++n) {
      float vv = bf2f(vt[n * 64 + d]);
      #pragma unroll
      for (int j = 0; j < 16; ++j)
        acc[j] += bf2f(kt[n * 64 + eg * 16 + j]) * vv;
    }
    if (t < 64) {
      #pragma unroll 8
      for (int n = 0; n < 32; ++n) ks += bf2f(kt[n * 64 + t]);
    }
  }
  #pragma unroll
  for (int j = 0; j < 16; ++j)
    kvt[((size_t)bh * 64 + d) * 64 + eg * 16 + j] = f2bf(acc[j]);
  if (t < 64) ksum[bh * 64 + t] = ks;
}

// ---------------------------------------------------------------------------
// Kernel 4: per (b,h): out[n][d] = (q[n] @ kv)[d] * 1/(q[n].ksum + eps)
//   128-row tiles, 4 waves each 32x64 (8 accumulators), K=64 (2 WMMA steps)
// ---------------------------------------------------------------------------
__global__ __launch_bounds__(128) void attn_kernel(
    const unsigned short* __restrict__ qb,
    const unsigned short* __restrict__ kvt,
    const float* __restrict__ ksum,
    unsigned short* __restrict__ attn) {
  __shared__ __align__(16) unsigned short At[128 * 64];
  __shared__ __align__(16) unsigned short Bt[64 * 64];
  __shared__ float zb[128];
  __shared__ float ksl[64];
  const int bh = blockIdx.y;
  const int bb = bh / HEADS, h = bh % HEADS;
  const int m0 = blockIdx.x * 128;
  const int t = threadIdx.x;
  const int wave = t >> 5, lane = t & 31;
  const int half = lane >> 4, lrow = lane & 15;

  {  // stage kv^T (B matrix, rows = d, K contiguous)
    int d = t >> 1, cg = (t & 1) * 32;
    const uint4* g = (const uint4*)(kvt + ((size_t)bh * 64 + d) * 64 + cg);
    uint4* s = (uint4*)(Bt + d * 64 + cg);
    s[0] = g[0]; s[1] = g[1]; s[2] = g[2]; s[3] = g[3];
  }
  if (t < 64) ksl[t] = ksum[bh * 64 + t];
  {  // stage q rows (zero-pad past NTOK)
    int n2 = m0 + t;
    uint4* s = (uint4*)(At + t * 64);
    if (n2 < NTOK) {
      const uint4* g = (const uint4*)(qb + ((size_t)bb * NTOK + n2) * C_DIM + h * EHEAD);
      #pragma unroll
      for (int i = 0; i < 8; ++i) s[i] = g[i];
    } else {
      uint4 z = {0, 0, 0, 0};
      #pragma unroll
      for (int i = 0; i < 8; ++i) s[i] = z;
    }
  }
  __syncthreads();
  {  // z = 1/(q . ksum + eps)
    float dot = 0.f;
    #pragma unroll 8
    for (int e = 0; e < 64; ++e) dot += bf2f(At[t * 64 + e]) * ksl[e];
    zb[t] = (m0 + t < NTOK) ? 1.0f / (dot + 1e-6f) : 0.0f;
  }
  __syncthreads();

  v8f acc[2][4] = {};
  #pragma unroll
  for (int kk = 0; kk < 64; kk += 32) {
    v16bf bfrag[4];
    #pragma unroll
    for (int nt = 0; nt < 4; ++nt) {
      const uint4* pb = (const uint4*)(Bt + (nt * 16 + lrow) * 64 + kk);
      bfrag[nt] = frag_cast(pb[half * 2], pb[half * 2 + 1]);
    }
    #pragma unroll
    for (int mt = 0; mt < 2; ++mt) {
      const uint4* pa = (const uint4*)(At + (wave * 32 + mt * 16 + lrow) * 64 + kk);
      v16bf afrag = frag_cast(pa[half * 2], pa[half * 2 + 1]);
      #pragma unroll
      for (int nt = 0; nt < 4; ++nt)
        acc[mt][nt] = __builtin_amdgcn_wmma_f32_16x16x32_bf16(
            false, afrag, false, bfrag[nt], (short)0, acc[mt][nt], false, false);
    }
  }

  #pragma unroll
  for (int mt = 0; mt < 2; ++mt) {
    int rbase = wave * 32 + mt * 16 + half * 8;
    #pragma unroll
    for (int nt = 0; nt < 4; ++nt) {
      int dcol = nt * 16 + lrow;
      v8f a = acc[mt][nt];
      #pragma unroll
      for (int r = 0; r < 8; ++r) {
        int rr = rbase + r;
        int n2 = m0 + rr;
        if (n2 < NTOK) {
          float v = a[r] * zb[rr];
          attn[((size_t)bb * NTOK + n2) * C_DIM + h * EHEAD + dcol] = f2bf(v);
        }
      }
    }
  }
}

// ---------------------------------------------------------------------------
// Kernel 5: out = attn @ Wproj^T + bproj  (M=25088, N=768, K=768), fp32 out
//   Same async double-buffered structure as gemm_qkv.
// ---------------------------------------------------------------------------
__global__ __launch_bounds__(128) void gemm_proj(
    const unsigned short* __restrict__ ab,
    const unsigned short* __restrict__ wb,
    const float* __restrict__ bias,
    float* __restrict__ out) {
  __shared__ __align__(16) unsigned short lA[2][128 * 32];
  __shared__ __align__(16) unsigned short lB[2][128 * 32];
  const int tid  = threadIdx.x;
  const int wave = tid >> 5, lane = tid & 31;
  const int wm = wave >> 1, wn = wave & 1;
  const int half = lane >> 4, lrow = lane & 15;
  const int m0 = blockIdx.x * 128;
  const int n0 = blockIdx.y * 128;

  const unsigned short* gA = ab + (size_t)(m0 + tid) * C_DIM;
  const unsigned short* gB = wb + (size_t)(n0 + tid) * C_DIM;
  const unsigned ldsA0 = lds_off(&lA[0][0]) + tid * 64;
  const unsigned ldsA1 = lds_off(&lA[1][0]) + tid * 64;
  const unsigned ldsB0 = lds_off(&lB[0][0]) + tid * 64;
  const unsigned ldsB1 = lds_off(&lB[1][0]) + tid * 64;

  v8f acc[4][4] = {};

  ASYNC_CP64(ldsA0, gA);
  ASYNC_CP64(ldsB0, gB);

  const int NK = C_DIM / 32;
  for (int i = 0; i < NK; ++i) {
    const int cur = i & 1;
    ASYNC_WAIT0();
    __syncthreads();
    if (i + 1 < NK) {
      const unsigned short* pa = gA + (i + 1) * 32;
      const unsigned short* pb = gB + (i + 1) * 32;
      ASYNC_CP64(cur ? ldsA0 : ldsA1, pa);
      ASYNC_CP64(cur ? ldsB0 : ldsB1, pb);
    }
    const unsigned short* bufA = lA[cur];
    const unsigned short* bufB = lB[cur];

    v16bf bfrag[4];
    #pragma unroll
    for (int nt = 0; nt < 4; ++nt) {
      const uint4* pb4 = (const uint4*)(bufB + (wn * 64 + nt * 16 + lrow) * 32);
      bfrag[nt] = frag_cast(pb4[half * 2], pb4[half * 2 + 1]);
    }
    #pragma unroll
    for (int mt = 0; mt < 4; ++mt) {
      const uint4* pa4 = (const uint4*)(bufA + (wm * 64 + mt * 16 + lrow) * 32);
      v16bf afrag = frag_cast(pa4[half * 2], pa4[half * 2 + 1]);
      #pragma unroll
      for (int nt = 0; nt < 4; ++nt)
        acc[mt][nt] = __builtin_amdgcn_wmma_f32_16x16x32_bf16(
            false, afrag, false, bfrag[nt], (short)0, acc[mt][nt], false, false);
    }
  }

  #pragma unroll
  for (int nt = 0; nt < 4; ++nt) {
    int c = n0 + wn * 64 + nt * 16 + lrow;
    float bv = bias[c];
    #pragma unroll
    for (int mt = 0; mt < 4; ++mt) {
      int mbase = m0 + wm * 64 + mt * 16 + half * 8;
      v8f a = acc[mt][nt];
      #pragma unroll
      for (int r = 0; r < 8; ++r)
        out[(size_t)(mbase + r) * C_DIM + c] = a[r] + bv;
    }
  }
}

// ---------------------------------------------------------------------------
// Host launch
// ---------------------------------------------------------------------------
extern "C" void kernel_launch(void* const* d_in, const int* in_sizes, int n_in,
                              void* d_out, int out_size, void* d_ws, size_t ws_size,
                              hipStream_t stream) {
  (void)in_sizes; (void)n_in; (void)out_size; (void)ws_size;
  const float* x     = (const float*)d_in[0];
  const float* Wqkv  = (const float*)d_in[1];
  const float* Wproj = (const float*)d_in[2];
  const float* bproj = (const float*)d_in[3];
  float* out = (float*)d_out;

  char* ws = (char*)d_ws;
  const size_t XB = (size_t)MROWS * C_DIM * 2;        // 38,535,168
  const size_t WQ = (size_t)3 * C_DIM * C_DIM * 2;    // 3,538,944
  const size_t WP = (size_t)C_DIM * C_DIM * 2;        // 1,179,648
  const size_t KVT = (size_t)192 * 64 * 64 * 2;       // 1,572,864
  unsigned short* xb     = (unsigned short*)ws;  ws += XB;
  unsigned short* wqkvb  = (unsigned short*)ws;  ws += WQ;
  unsigned short* wprojb = (unsigned short*)ws;  ws += WP;
  unsigned short* qb     = (unsigned short*)ws;  ws += XB;
  unsigned short* kb     = (unsigned short*)ws;  ws += XB;
  unsigned short* vb     = (unsigned short*)ws;  ws += XB;
  unsigned short* kvt    = (unsigned short*)ws;  ws += KVT;
  float*          ksum   = (float*)ws;           ws += 192 * 64 * 4;
  unsigned short* attn   = xb;   // x is dead after gemm_qkv; reuse its slot

  cvt_f32_bf16<<<4096, 256, 0, stream>>>(x,     xb,     (long long)MROWS * C_DIM);
  cvt_f32_bf16<<<1024, 256, 0, stream>>>(Wqkv,  wqkvb,  (long long)3 * C_DIM * C_DIM);
  cvt_f32_bf16<<<512,  256, 0, stream>>>(Wproj, wprojb, (long long)C_DIM * C_DIM);

  gemm_qkv<<<dim3(MROWS / 128, (3 * C_DIM) / 128), 128, 0, stream>>>(xb, wqkvb, qb, kb, vb);
  kv_kernel<<<192, 256, 0, stream>>>(kb, vb, kvt, ksum);
  attn_kernel<<<dim3((NTOK + 127) / 128, 192), 128, 0, stream>>>(qb, kvt, ksum, attn);
  gemm_proj<<<dim3(MROWS / 128, C_DIM / 128), 128, 0, stream>>>(attn, wprojb, bproj, out);
}
